// SceneNetStereoJoint_17111149707878
// MI455X (gfx1250) — matrix operand order repaired and checked by reading, not compiled, not run on hardware
//
#include <hip/hip_runtime.h>
#include <cmath>

// Problem constants (fixed by reference setup_inputs)
#define Bn 4
#define Cn 64
#define Hn 128
#define Wn 416
#define HW (Hn * Wn)
#define NCH 273
#define MDr 4
#define NEG_SLOPE 0.1f

typedef _Float16 half_t;
typedef __attribute__((ext_vector_type(16))) _Float16 v16h;
typedef __attribute__((ext_vector_type(8)))  float    v8f;
typedef __attribute__((ext_vector_type(4)))  unsigned int v4u;
typedef __attribute__((ext_vector_type(4)))  int          v4i;

// LDS geometry: 64 channels + 8 pad halfs per pixel (144B pitch -> conflict-free b128 frag reads)
#define APAD 72
#define SA_BYTES (Wn * APAD * 2)          // 59904  : x1 row, [w][ch] f16
#define SB9_X 432                          // 9x9 corr: x in [-4, 427]
#define SB9_BYTES (SB9_X * APAD * 2)       // 62208
#define SBD_X 608                          // disp corr: x in [-96, 511]
#define SBD_BYTES (SBD_X * APAD * 2)       // 87552
#define SG9_STRIDE 33                      // 16x32 G tile + pad
#define SGD_STRIDE 209                     // 16x208 G row + pad

#define AS1 __attribute__((address_space(1)))
#define AS3 __attribute__((address_space(3)))

// ---- CDNA5 async global->LDS path (ASYNCcnt), guarded so fallback still compiles ----
#if defined(__has_builtin)
#if __has_builtin(__builtin_amdgcn_global_load_async_to_lds_b128) && \
    __has_builtin(__builtin_amdgcn_s_wait_asynccnt)
#define USE_ASYNC_LDS 1
#endif
#endif
#ifndef USE_ASYNC_LDS
#define USE_ASYNC_LDS 0
#endif

__device__ __forceinline__ void lds_copy16(unsigned short* dst, const v4u* src) {
#if USE_ASYNC_LDS
  // global_load_async_to_lds_b128: bytes flow HBM/L2 -> LDS without touching VGPRs/VALU.
  // Param types per hipcc diagnostic: (v4i __device__ *, v4i __shared__ *, imm offset, imm cpol)
  __builtin_amdgcn_global_load_async_to_lds_b128(
      (AS1 v4i*)(unsigned long long)(size_t)src,
      (AS3 v4i*)(unsigned int)(size_t)dst, 0, 0);
#else
  *(v4u*)dst = *src;
#endif
}
__device__ __forceinline__ void lds_fill_fence() {
#if USE_ASYNC_LDS
  __builtin_amdgcn_s_wait_asynccnt(0);   // wave's async fills done before it signals the barrier
#endif
}

__device__ __forceinline__ float lrelu(float v) { return v >= 0.f ? v : NEG_SLOPE * v; }

// A fragment (16x32 f16, M=pixels, K=channels). Lane L: M = L%16.
// v16h halves 0..7 -> K = (L<16 ? 0..7 : 8..15)+32*kh ; halves 8..15 -> +16.
__device__ __forceinline__ v16h lds_a_frag(const unsigned short* sA, int w, int hi, int kh) {
  const unsigned short* p = sA + w * APAD + kh * 32 + hi * 8;
  union { v16h v; v4u q[2]; } u;
  u.q[0] = *(const v4u*)(p);
  u.q[1] = *(const v4u*)(p + 16);
  return u.v;
}
// B fragment (32x16 f16, K=channels, N=j). Lane L: N = L%16.
// halves 0..7 -> K = (L<16 ? 0..7 : 8..15)+32*kh ; halves 8..15 -> +16.
__device__ __forceinline__ v16h lds_b_frag(const unsigned short* sB, int xl, int hi, int kh) {
  const unsigned short* p = sB + xl * APAD + kh * 32 + hi * 8;
  union { v16h v; v4u q[2]; } u;
  u.q[0] = *(const v4u*)(p);
  u.q[1] = *(const v4u*)(p + 16);
  return u.v;
}

// ---------------- NCHW fp32 -> NHWC f16 convert ----------------
__global__ void cvt_kernel(const float* __restrict__ src, half_t* __restrict__ dst) {
  int pix = blockIdx.x * blockDim.x + threadIdx.x;
  if (pix >= Bn * HW) return;
  int b = pix / HW, p = pix - b * HW;
  const float* s = src + (size_t)b * Cn * HW + p;
  half_t* d = dst + (size_t)pix * Cn;
#pragma unroll 8
  for (int c = 0; c < Cn; ++c) d[c] = (half_t)s[(size_t)c * HW];
}

// ---------------- bilinear warp (zeros padding) + f16 NHWC ----------------
__global__ void warp_kernel(const float* __restrict__ x2, const float* __restrict__ flow,
                            half_t* __restrict__ dst) {
  int pix = blockIdx.x * blockDim.x + threadIdx.x;
  if (pix >= Bn * HW) return;
  int b = pix / HW, p = pix - b * HW;
  int h = p / Wn, w = p - h * Wn;
  float gx = (float)w + flow[((size_t)(b * 2 + 0) * Hn + h) * Wn + w];
  float gy = (float)h + flow[((size_t)(b * 2 + 1) * Hn + h) * Wn + w];
  float x0f = floorf(gx), y0f = floorf(gy);
  float wx = gx - x0f, wy = gy - y0f;
  int x0 = (int)x0f, y0 = (int)y0f;
  int x1i = x0 + 1, y1i = y0 + 1;
  float vx0 = (x0 >= 0 && x0 < Wn) ? 1.f : 0.f;
  float vx1 = (x1i >= 0 && x1i < Wn) ? 1.f : 0.f;
  float vy0 = (y0 >= 0 && y0 < Hn) ? 1.f : 0.f;
  float vy1 = (y1i >= 0 && y1i < Hn) ? 1.f : 0.f;
  int xc0 = min(max(x0, 0), Wn - 1), xc1 = min(max(x1i, 0), Wn - 1);
  int yc0 = min(max(y0, 0), Hn - 1), yc1 = min(max(y1i, 0), Hn - 1);
  float w00 = (1.f - wx) * (1.f - wy) * vx0 * vy0;
  float w01 = wx * (1.f - wy) * vx1 * vy0;
  float w10 = (1.f - wx) * wy * vx0 * vy1;
  float w11 = wx * wy * vx1 * vy1;
  size_t o00 = (size_t)yc0 * Wn + xc0, o01 = (size_t)yc0 * Wn + xc1;
  size_t o10 = (size_t)yc1 * Wn + xc0, o11 = (size_t)yc1 * Wn + xc1;
  const float* img = x2 + (size_t)b * Cn * HW;
  half_t* d = dst + (size_t)pix * Cn;
#pragma unroll 4
  for (int c = 0; c < Cn; ++c) {
    size_t cb = (size_t)c * HW;
    float v = w00 * img[cb + o00] + w01 * img[cb + o01] +
              w10 * img[cb + o10] + w11 * img[cb + o11];
    d[c] = (half_t)v;
  }
}

// ---------------- 9x9 cost volume -> out channels [0,81) ----------------
__global__ __launch_bounds__(256) void corr9_kernel(const half_t* __restrict__ x1h,
                                                    const half_t* __restrict__ x2wh,
                                                    float* __restrict__ out) {
  const int bh = blockIdx.x;
  const int b = bh / Hn, h = bh - b * Hn;
  const int tid = threadIdx.x, lane = tid & 31, wave = tid >> 5;
  const int m = lane & 15, hi = lane >> 4;
  extern __shared__ char smem[];
  unsigned short* sA = (unsigned short*)smem;
  unsigned short* sB = (unsigned short*)(smem + SA_BYTES);
  float* sG = (float*)(smem + SA_BYTES + SB9_BYTES) + wave * 16 * SG9_STRIDE;
  const v4u* x1q = (const v4u*)x1h;
  const v4u* x2q = (const v4u*)x2wh;
  // stage x1 row: [w][ch] f16, 16B chunks
  for (int q = tid; q < Wn * 8; q += 256) {
    int w = q >> 3, cc = q & 7;
    lds_copy16(sA + w * APAD + cc * 8, x1q + (size_t)((b * Hn + h) * Wn + w) * 8 + cc);
  }
  lds_fill_fence();
  for (int dy = -MDr; dy <= MDr; ++dy) {
    __syncthreads();
    int y = h + dy;
    bool rowok = (y >= 0 && y < Hn);
    for (int q = tid; q < SB9_X * 8; q += 256) {
      int xl = q >> 3, cc = q & 7;
      int x = xl - MDr;
      unsigned short* dst = sB + xl * APAD + cc * 8;
      if (rowok && x >= 0 && x < Wn) {
        lds_copy16(dst, x2q + (size_t)((b * Hn + y) * Wn + x) * 8 + cc);
      } else {
        v4u z = {};
        *(v4u*)dst = z;
      }
    }
    lds_fill_fence();
    __syncthreads();
    for (int tile = wave; tile < 26; tile += 8) {
      int w0 = tile * 16;
      v16h a0 = lds_a_frag(sA, w0 + m, hi, 0);
      v16h a1 = lds_a_frag(sA, w0 + m, hi, 1);
#pragma unroll
      for (int t = 0; t < 2; ++t) {
        // G[m][j] = sum_k x1[k, w0+m] * x2w[k, w0 + 16t + n - 4] ; out[m,dx] = G[m, m+dx+4]
        v16h b0 = lds_b_frag(sB, w0 + 16 * t + m, hi, 0);
        v16h b1 = lds_b_frag(sB, w0 + 16 * t + m, hi, 1);
        v8f acc = {};
        acc = __builtin_amdgcn_wmma_f32_16x16x32_f16(false, a0, false, b0, (short)0, acc, false, false);
        acc = __builtin_amdgcn_wmma_f32_16x16x32_f16(false, a1, false, b1, (short)0, acc, false, false);
#pragma unroll
        for (int r = 0; r < 8; ++r) sG[(r + 8 * hi) * SG9_STRIDE + 16 * t + m] = acc[r];
      }
      // extract diagonal band, coalesced stores (2 channels x 16 pixels per iter)
#pragma unroll
      for (int it = 0; it < 5; ++it) {
        int ch9 = 2 * it + hi;
        if (ch9 < 9) {
          float v = lrelu(sG[m * SG9_STRIDE + m + ch9]);
          int outc = (dy + MDr) * 9 + ch9;
          out[(((size_t)b * NCH + outc) * Hn + h) * Wn + w0 + m] = v;
        }
      }
    }
  }
}

// ---------------- 192-disp cost volume -> out channels [81,273) ----------------
__global__ __launch_bounds__(256) void dispcorr_kernel(const half_t* __restrict__ x1h,
                                                       const half_t* __restrict__ r1h,
                                                       float* __restrict__ out) {
  const int bh = blockIdx.x;
  const int b = bh / Hn, h = bh - b * Hn;
  const int tid = threadIdx.x, lane = tid & 31, wave = tid >> 5;
  const int m = lane & 15, hi = lane >> 4;
  extern __shared__ char smem[];
  unsigned short* sA = (unsigned short*)smem;
  unsigned short* sB = (unsigned short*)(smem + SA_BYTES);
  float* sG = (float*)(smem + SA_BYTES + SBD_BYTES) + wave * 16 * SGD_STRIDE;
  const v4u* x1q = (const v4u*)x1h;
  const v4u* r1q = (const v4u*)r1h;
  for (int q = tid; q < Wn * 8; q += 256) {
    int w = q >> 3, cc = q & 7;
    lds_copy16(sA + w * APAD + cc * 8, x1q + (size_t)((b * Hn + h) * Wn + w) * 8 + cc);
  }
  for (int q = tid; q < SBD_X * 8; q += 256) {
    int xl = q >> 3, cc = q & 7;
    int x = xl - 96;                       // sB covers x in [-96, 511], zero-padded
    unsigned short* dst = sB + xl * APAD + cc * 8;
    if (x >= 0 && x < Wn) {
      lds_copy16(dst, r1q + (size_t)((b * Hn + h) * Wn + x) * 8 + cc);
    } else {
      v4u z = {};
      *(v4u*)dst = z;
    }
  }
  lds_fill_fence();
  __syncthreads();
  for (int tile = wave; tile < 26; tile += 8) {
    int w0 = tile * 16;
    v16h a0 = lds_a_frag(sA, w0 + m, hi, 0);
    v16h a1 = lds_a_frag(sA, w0 + m, hi, 1);
    // G[m][jl] = sum_k x1[k, w0+m] * r1[k, w0 - 96 + jl], jl in [0,208); out ch = jl - m
    for (int t = 0; t < 13; ++t) {
      v16h b0 = lds_b_frag(sB, w0 + 16 * t + m, hi, 0);
      v16h b1 = lds_b_frag(sB, w0 + 16 * t + m, hi, 1);
      v8f acc = {};
      acc = __builtin_amdgcn_wmma_f32_16x16x32_f16(false, a0, false, b0, (short)0, acc, false, false);
      acc = __builtin_amdgcn_wmma_f32_16x16x32_f16(false, a1, false, b1, (short)0, acc, false, false);
#pragma unroll
      for (int r = 0; r < 8; ++r) sG[(r + 8 * hi) * SGD_STRIDE + 16 * t + m] = acc[r];
    }
    // 192 channels: diagonal extract, coalesced (2 ch x 16 px per iter)
    for (int it = 0; it < 96; ++it) {
      int ch = 2 * it + hi;
      float v = lrelu(sG[m * SGD_STRIDE + m + ch]);
      out[(((size_t)b * NCH + 81 + ch) * Hn + h) * Wn + w0 + m] = v;
    }
  }
}

extern "C" void kernel_launch(void* const* d_in, const int* in_sizes, int n_in,
                              void* d_out, int out_size, void* d_ws, size_t ws_size,
                              hipStream_t stream) {
  (void)in_sizes; (void)n_in; (void)out_size; (void)ws_size;
  const float* x1   = (const float*)d_in[0];
  const float* x2   = (const float*)d_in[1];
  const float* r1   = (const float*)d_in[2];
  const float* flow = (const float*)d_in[3];
  float* out = (float*)d_out;

  const size_t planeElems = (size_t)Bn * HW * Cn;   // f16 elements per tensor
  half_t* x1h  = (half_t*)d_ws;
  half_t* r1h  = x1h + planeElems;
  half_t* x2wh = r1h + planeElems;

  const int npix = Bn * HW;
  const int cvtBlocks = (npix + 255) / 256;
  cvt_kernel<<<cvtBlocks, 256, 0, stream>>>(x1, x1h);
  cvt_kernel<<<cvtBlocks, 256, 0, stream>>>(r1, r1h);
  warp_kernel<<<cvtBlocks, 256, 0, stream>>>(x2, flow, x2wh);

  const int rows = Bn * Hn;  // one workgroup per (b, h) row
  const size_t lds9 = SA_BYTES + SB9_BYTES + (size_t)8 * 16 * SG9_STRIDE * 4;   // ~139 KB
  const size_t ldsD = SA_BYTES + SBD_BYTES + (size_t)8 * 16 * SGD_STRIDE * 4;   // ~249 KB (<320KB WGP LDS)
  corr9_kernel<<<rows, 256, lds9, stream>>>(x1h, x2wh, out);
  dispcorr_kernel<<<rows, 256, ldsD, stream>>>(x1h, r1h, out);
}
